// MaskedMultiHeadAttention_45595372814614
// MI455X (gfx1250) — compile-verified, matrix operand order checked by reading, not run on hardware
//
#include <hip/hip_runtime.h>

// ---------------------------------------------------------------------------
// Masked multi-head attention forward for MI455X (gfx1250, wave32, WMMA).
// fp32->bf16 convert -> QKV projections (WMMA GEMM, 32x64 wave tiles) ->
// causal flash-attention (WMMA QK^T + PV, online softmax, V tile staged via
// Tensor Data Mover) -> output projection. Intermediates live in d_ws as
// bf16; the whole working set (~126 MB) fits in the 192 MB L2.
// ---------------------------------------------------------------------------

#define DEV __device__ __forceinline__

constexpr int BB = 4, SS = 2048, EE = 1024, HH = 16, DD = 64;
constexpr int MM = BB * SS;  // 8192 token rows

typedef __attribute__((ext_vector_type(16))) __bf16 v16bf;
typedef __attribute__((ext_vector_type(8)))  float  v8f;
typedef __attribute__((ext_vector_type(4)))  unsigned int u32x4;
typedef __attribute__((ext_vector_type(8)))  int  i32x8;
typedef __attribute__((ext_vector_type(4)))  int  i32x4;

#if defined(__has_builtin)
#if __has_builtin(__builtin_amdgcn_tensor_load_to_lds) && \
    __has_builtin(__builtin_amdgcn_s_wait_tensorcnt)
#define USE_TDM 1
#endif
#endif

union Frag16 {
    v16bf bf;
    u32x4 q[2];
    unsigned short h[16];
};

DEV unsigned short f32_to_bf16(float f) {
    unsigned int u = __float_as_uint(f);
    unsigned int r = u + 0x7fffu + ((u >> 16) & 1u);  // round-to-nearest-even
    return (unsigned short)(r >> 16);
}

DEV v8f wmma_bf16(v16bf a, v16bf b, v8f c) {
    // D(16x16 f32) = A(16x32 bf16) * B(32x16 bf16) + C
    return __builtin_amdgcn_wmma_f32_16x16x32_bf16(
        false, a, false, b, (short)0, c, false, false);
}

// Row-major fragment gather for the 16-bit 16x32 A layout (and the symmetric
// 32x16 B layout when B is a transposed row-major matrix):
//   lanes 0-15 : K = {k0+0..7,  k0+16..23}
//   lanes 16-31: K = {k0+8..15, k0+24..31}
// `p` must already point at (row, k0 + half*8); two contiguous 16B chunks.
DEV v16bf load_frag_row(const unsigned short* p) {
    Frag16 f;
    f.q[0] = *reinterpret_cast<const u32x4*>(p);
    f.q[1] = *reinterpret_cast<const u32x4*>(p + 16);
    return f.bf;
}

// ---------------------------------------------------------------------------
// Kernel 1: fp32 -> bf16 elementwise conversion (4 elems / thread)
// ---------------------------------------------------------------------------
__global__ __launch_bounds__(256) void mha_cvt_bf16(
    const float* __restrict__ in, unsigned short* __restrict__ out, int n) {
    int i = (blockIdx.x * blockDim.x + threadIdx.x) * 4;
    if (i + 3 < n) {
        float4 v = *reinterpret_cast<const float4*>(in + i);
        ushort4 o;
        o.x = f32_to_bf16(v.x);
        o.y = f32_to_bf16(v.y);
        o.z = f32_to_bf16(v.z);
        o.w = f32_to_bf16(v.w);
        *reinterpret_cast<ushort4*>(out + i) = o;
    }
}

// ---------------------------------------------------------------------------
// GEMM wave tile: C(32 x 64) = X(32 x K) * W^T(K x 64), K = 1024.
// Two A-row fragments share each B fragment -> ~21 FLOP/byte of cache
// traffic. Because B = W^T, lane n's B-fragment column is row n of W, so
// both operands use the same coalesced row gather.
// ---------------------------------------------------------------------------
DEV void gemm_wave_tile32(const unsigned short* __restrict__ X,
                          const unsigned short* __restrict__ W,
                          int m0, int n0, v8f acc[8], int lr, int half) {
    const unsigned short* arow0 = X + (size_t)(m0 + lr) * EE + half * 8;
    const unsigned short* arow1 = arow0 + (size_t)16 * EE;
    const unsigned short* brow0 = W + (size_t)(n0 + 0 * 16 + lr) * EE + half * 8;
    const unsigned short* brow1 = W + (size_t)(n0 + 1 * 16 + lr) * EE + half * 8;
    const unsigned short* brow2 = W + (size_t)(n0 + 2 * 16 + lr) * EE + half * 8;
    const unsigned short* brow3 = W + (size_t)(n0 + 3 * 16 + lr) * EE + half * 8;
#pragma unroll 2
    for (int k0 = 0; k0 < EE; k0 += 32) {
        __builtin_prefetch(arow0 + k0 + 256, 0, 3);  // global_prefetch
        __builtin_prefetch(arow1 + k0 + 256, 0, 3);
        v16bf a0 = load_frag_row(arow0 + k0);
        v16bf a1 = load_frag_row(arow1 + k0);
        v16bf b0 = load_frag_row(brow0 + k0);
        acc[0] = wmma_bf16(a0, b0, acc[0]);
        acc[4] = wmma_bf16(a1, b0, acc[4]);
        v16bf b1 = load_frag_row(brow1 + k0);
        acc[1] = wmma_bf16(a0, b1, acc[1]);
        acc[5] = wmma_bf16(a1, b1, acc[5]);
        v16bf b2 = load_frag_row(brow2 + k0);
        acc[2] = wmma_bf16(a0, b2, acc[2]);
        acc[6] = wmma_bf16(a1, b2, acc[6]);
        v16bf b3 = load_frag_row(brow3 + k0);
        acc[3] = wmma_bf16(a0, b3, acc[3]);
        acc[7] = wmma_bf16(a1, b3, acc[7]);
    }
}

// ---------------------------------------------------------------------------
// Kernel 2: QKV projection. out[b,h,s,d] = bf16((X @ W^T + bias) * scale)
// One wave -> 32 rows x 64 cols (= exactly one head slice).
// ---------------------------------------------------------------------------
__global__ __launch_bounds__(256) void mha_proj_qkv(
    const unsigned short* __restrict__ X, const unsigned short* __restrict__ W,
    const float* __restrict__ bias, unsigned short* __restrict__ outp,
    float scale) {
    const int lane = threadIdx.x & 31, lr = lane & 15, half = lane >> 4;
    const int wid = blockIdx.x * 8 + (threadIdx.x >> 5);
    const int tilesN = EE / 64;                 // 16
    const int m0 = (wid / tilesN) * 32;
    const int n0 = (wid % tilesN) * 64;

    v8f acc[8] = {};
    gemm_wave_tile32(X, W, m0, n0, acc, lr, half);

    const int h = n0 >> 6;  // 64 cols == one head
#pragma unroll
    for (int t = 0; t < 4; ++t) {
        const int n = n0 + t * 16 + lr;
        const float bn = bias[n];
#pragma unroll
        for (int g = 0; g < 2; ++g)  // two 16-row groups
#pragma unroll
            for (int r = 0; r < 8; ++r) {
                const int m = m0 + g * 16 + r + half * 8;
                const int b = m >> 11, s = m & (SS - 1);
                const size_t idx =
                    (((size_t)b * HH + h) * SS + s) * DD + t * 16 + lr;
                outp[idx] = f32_to_bf16((acc[g * 4 + t][r] + bn) * scale);
            }
    }
}

// ---------------------------------------------------------------------------
// Kernel 3: causal flash attention. One wave per 16-query tile of one head.
// Key tiles of 32; scores/context via WMMA; online softmax. V tiles are
// staged into LDS by the Tensor Data Mover (2D descriptor with LDS padding
// to a 72-half pitch), overlapping the DMA with the QK^T WMMAs.
// ---------------------------------------------------------------------------
__global__ __launch_bounds__(256) void mha_attn(
    const unsigned short* __restrict__ Qp, const unsigned short* __restrict__ Kp,
    const unsigned short* __restrict__ Vp, unsigned short* __restrict__ ctx) {
    // per-wave LDS slice: P tile 16x(32 pad 40) halves + V tile 32x(64 pad 72)
    __shared__ unsigned short smem[8 * 2944];
    const int widb = threadIdx.x >> 5;
    unsigned short* Pl = smem + widb * 2944;   // 640 halves
    unsigned short* Vl = Pl + 640;             // 2304 halves

    const int wid = blockIdx.x * 8 + widb;
    const int qt = wid & 127, bh = wid >> 7;
    const int h = bh & (HH - 1), b = bh >> 4;
    const int q0 = qt * 16;
    const int lane = threadIdx.x & 31, lr = lane & 15, half = lane >> 4;

    const size_t headoff = ((size_t)(b * HH + h)) * SS * DD;
    const unsigned short* Qh = Qp + headoff;
    const unsigned short* Kh = Kp + headoff;
    const unsigned short* Vh = Vp + headoff;

#if defined(USE_TDM)
    // LDS byte offset of this wave's V tile (workgroup-relative).
    typedef __attribute__((address_space(3))) const unsigned short* lds_usp;
    const unsigned vl_lds_byte = (unsigned)(unsigned long long)(lds_usp)Vl;
    // D# group1: data_size=2B, pad_enable, pad_interval=32 DW, pad_amount=4 DW,
    // tensor_dim0=64, tensor_dim1=32(tile rows), tile_dim0=64, tile_dim1=32,
    // tensor_dim0_stride=64.
    const i32x8 tdm_g1 = {
        (int)((1u << 16) | (1u << 20) | (4u << 22) | (3u << 25)),
        (int)(64u << 16),   // tensor_dim0[15:0]
        (int)(32u << 16),   // tensor_dim0[31:16]=0 | tensor_dim1[15:0]
        (int)(64u << 16),   // tensor_dim1[31:16]=0 | tile_dim0
        (int)(32u),         // tile_dim1 | tile_dim2=0
        (int)(64u),         // tensor_dim0_stride[31:0]
        0, 0};
    const i32x4 tdm_zero4 = {0, 0, 0, 0};
    const i32x8 tdm_zero8 = {0, 0, 0, 0, 0, 0, 0, 0};
#endif

    // Q fragments for this tile (d = 0..31 and 32..63); scale pre-folded.
    const unsigned short* qrow = Qh + (size_t)(q0 + lr) * DD + half * 8;
    v16bf qf0 = load_frag_row(qrow);
    v16bf qf1 = load_frag_row(qrow + 32);

    v8f acc[4] = {};          // context 16x64 (f32)
    float mrow[8], lrow[8];
#pragma unroll
    for (int r = 0; r < 8; ++r) { mrow[r] = -3.0e38f; lrow[r] = 0.0f; }

    const int ktEnd = q0 + 16;  // causal bound (exclusive key index)
    for (int kt0 = 0; kt0 < ktEnd; kt0 += 32) {
        // previous iteration's LDS reads must finish before V is overwritten
        asm volatile("s_wait_dscnt 0x0" ::: "memory");

        // stage V tile (32 keys x 64 d) into LDS
#if defined(USE_TDM)
        {
            const unsigned long long ga =
                (unsigned long long)(Vh + (size_t)kt0 * DD);
            u32x4 g0;
            g0[0] = 1u;  // count=1, user descriptor
            g0[1] = (unsigned)__builtin_amdgcn_readfirstlane((int)vl_lds_byte);
            g0[2] = (unsigned)__builtin_amdgcn_readfirstlane((int)(ga & 0xffffffffu));
            g0[3] = (unsigned)__builtin_amdgcn_readfirstlane(
                        (int)(((ga >> 32) & 0x01ffffffu) | 0x80000000u));  // type=2
            __builtin_amdgcn_tensor_load_to_lds(g0, tdm_g1, tdm_zero4,
                                                tdm_zero4, tdm_zero8, 0);
        }
#else
        {
            const u32x4* vsrc =
                reinterpret_cast<const u32x4*>(Vh + (size_t)(kt0 + lane) * DD);
            u32x4* vdst = reinterpret_cast<u32x4*>(Vl + lane * 72);
#pragma unroll
            for (int j = 0; j < 4; ++j) vdst[j] = vsrc[j];
        }
#endif

        // scores S(16x32) = Q(16x64) * K^T(64x32) -- overlaps the V DMA
        v8f sc[2] = {};
#pragma unroll
        for (int nt = 0; nt < 2; ++nt) {
            const unsigned short* krow =
                Kh + (size_t)(kt0 + nt * 16 + lr) * DD + half * 8;
            sc[nt] = wmma_bf16(qf0, load_frag_row(krow), sc[nt]);
            sc[nt] = wmma_bf16(qf1, load_frag_row(krow + 32), sc[nt]);
        }

        // causal mask (only tiles straddling the diagonal)
        if (kt0 + 31 > q0) {
#pragma unroll
            for (int nt = 0; nt < 2; ++nt)
#pragma unroll
                for (int r = 0; r < 8; ++r) {
                    const int key = kt0 + nt * 16 + lr;
                    const int qi = q0 + r + half * 8;
                    if (key > qi) sc[nt][r] = -3.0e38f;
                }
        }

        // online softmax: rows live across 16-lane half-waves
#pragma unroll
        for (int r = 0; r < 8; ++r) {
            float mx = fmaxf(sc[0][r], sc[1][r]);
#pragma unroll
            for (int off = 1; off < 16; off <<= 1)
                mx = fmaxf(mx, __shfl_xor(mx, off, 16));
            const float mnew = fmaxf(mrow[r], mx);
            const float corr = __expf(mrow[r] - mnew);
            mrow[r] = mnew;
            const float p0 = __expf(sc[0][r] - mnew);
            const float p1 = __expf(sc[1][r] - mnew);
            sc[0][r] = p0;
            sc[1][r] = p1;
            float rs = p0 + p1;
#pragma unroll
            for (int off = 1; off < 16; off <<= 1)
                rs += __shfl_xor(rs, off, 16);
            lrow[r] = lrow[r] * corr + rs;
#pragma unroll
            for (int t = 0; t < 4; ++t) acc[t][r] *= corr;
        }

        // reshape P: f32 C-layout -> bf16 A-fragment via LDS
#pragma unroll
        for (int nt = 0; nt < 2; ++nt)
#pragma unroll
            for (int r = 0; r < 8; ++r)
                Pl[(r + half * 8) * 40 + nt * 16 + lr] = f32_to_bf16(sc[nt][r]);

        __builtin_amdgcn_wave_barrier();
        asm volatile("s_wait_dscnt 0x0" ::: "memory");  // P stores visible
#if defined(USE_TDM)
        __builtin_amdgcn_s_wait_tensorcnt(0);            // V tile landed
#endif

        v16bf pf = load_frag_row(Pl + lr * 40 + half * 8);

        // context += P(16x32) * V(32x64); V B-frags gathered column-wise
#pragma unroll
        for (int t = 0; t < 4; ++t) {
            Frag16 vf;
            const int dn = t * 16 + lr;
#pragma unroll
            for (int j = 0; j < 8; ++j) {
                const int kk = (j < 4 ? 2 * j : 16 + 2 * (j - 4)) + half * 8;
                vf.h[2 * j]     = Vl[kk * 72 + dn];
                vf.h[2 * j + 1] = Vl[(kk + 1) * 72 + dn];
            }
            acc[t] = wmma_bf16(pf, vf.bf, acc[t]);
        }
        __builtin_amdgcn_wave_barrier();
    }

    // normalize and write context back to [B,S,E] bf16
    float invl[8];
#pragma unroll
    for (int r = 0; r < 8; ++r) invl[r] = 1.0f / lrow[r];
#pragma unroll
    for (int t = 0; t < 4; ++t)
#pragma unroll
        for (int r = 0; r < 8; ++r) {
            const size_t row = (size_t)b * SS + q0 + r + half * 8;
            ctx[row * EE + h * DD + t * 16 + lr] =
                f32_to_bf16(acc[t][r] * invl[r]);
        }
}

// ---------------------------------------------------------------------------
// Kernel 4: output projection. out_f32 = ctx @ Wo^T + bo
// ---------------------------------------------------------------------------
__global__ __launch_bounds__(256) void mha_proj_out(
    const unsigned short* __restrict__ X, const unsigned short* __restrict__ W,
    const float* __restrict__ bias, float* __restrict__ out) {
    const int lane = threadIdx.x & 31, lr = lane & 15, half = lane >> 4;
    const int wid = blockIdx.x * 8 + (threadIdx.x >> 5);
    const int tilesN = EE / 64;
    const int m0 = (wid / tilesN) * 32;
    const int n0 = (wid % tilesN) * 64;

    v8f acc[8] = {};
    gemm_wave_tile32(X, W, m0, n0, acc, lr, half);

#pragma unroll
    for (int t = 0; t < 4; ++t) {
        const int n = n0 + t * 16 + lr;
        const float bn = bias[n];
#pragma unroll
        for (int g = 0; g < 2; ++g)
#pragma unroll
            for (int r = 0; r < 8; ++r) {
                const int m = m0 + g * 16 + r + half * 8;
                out[(size_t)m * EE + n] = acc[g * 4 + t][r] + bn;
            }
    }
}

// ---------------------------------------------------------------------------
// Host launch
// ---------------------------------------------------------------------------
extern "C" void kernel_launch(void* const* d_in, const int* in_sizes, int n_in,
                              void* d_out, int out_size, void* d_ws,
                              size_t ws_size, hipStream_t stream) {
    (void)in_sizes; (void)n_in; (void)out_size; (void)ws_size;
    const float* q    = (const float*)d_in[0];
    const float* k    = (const float*)d_in[1];
    const float* v    = (const float*)d_in[2];
    // d_in[3] = mask: causal, handled analytically
    const float* Wq = (const float*)d_in[4];
    const float* bq = (const float*)d_in[5];
    const float* Wk = (const float*)d_in[6];
    const float* bk = (const float*)d_in[7];
    const float* Wv = (const float*)d_in[8];
    const float* bv = (const float*)d_in[9];
    const float* Wo = (const float*)d_in[10];
    const float* bo = (const float*)d_in[11];
    float* out = (float*)d_out;

    const size_t XN = (size_t)MM * EE;  // 8388608 elems
    const size_t WN = (size_t)EE * EE;  // 1048576 elems
    unsigned short* ws = (unsigned short*)d_ws;
    unsigned short* qx  = ws;
    unsigned short* kx  = qx + XN;
    unsigned short* vx  = kx + XN;
    unsigned short* wqb = vx + XN;
    unsigned short* wkb = wqb + WN;
    unsigned short* wvb = wkb + WN;
    unsigned short* wob = wvb + WN;
    unsigned short* Qp  = wob + WN;   // [B,H,S,D]
    unsigned short* Kp  = Qp + XN;
    unsigned short* Vp  = Kp + XN;
    unsigned short* ctx = Vp + XN;    // [B,S,E]
    // total ws use: (7*XN + 4*WN) * 2 bytes = ~126 MB

    const dim3 blk(256);
    // fp32 -> bf16 staging
    mha_cvt_bf16<<<dim3((unsigned)(XN / 1024)), blk, 0, stream>>>(q, qx, (int)XN);
    mha_cvt_bf16<<<dim3((unsigned)(XN / 1024)), blk, 0, stream>>>(k, kx, (int)XN);
    mha_cvt_bf16<<<dim3((unsigned)(XN / 1024)), blk, 0, stream>>>(v, vx, (int)XN);
    mha_cvt_bf16<<<dim3((unsigned)(WN / 1024)), blk, 0, stream>>>(Wq, wqb, (int)WN);
    mha_cvt_bf16<<<dim3((unsigned)(WN / 1024)), blk, 0, stream>>>(Wk, wkb, (int)WN);
    mha_cvt_bf16<<<dim3((unsigned)(WN / 1024)), blk, 0, stream>>>(Wv, wvb, (int)WN);
    mha_cvt_bf16<<<dim3((unsigned)(WN / 1024)), blk, 0, stream>>>(Wo, wob, (int)WN);

    // projections: (8192/32)*(1024/64) = 4096 wave tiles / 8 waves = 512 blocks
    const dim3 pgrid(512);
    mha_proj_qkv<<<pgrid, blk, 0, stream>>>(qx, wqb, bq, Qp, 0.125f);  // 1/sqrt(64)
    mha_proj_qkv<<<pgrid, blk, 0, stream>>>(kx, wkb, bk, Kp, 1.0f);
    mha_proj_qkv<<<pgrid, blk, 0, stream>>>(vx, wvb, bv, Vp, 1.0f);

    // attention: B*H*(S/16) = 8192 wave tiles -> 1024 blocks
    mha_attn<<<dim3(1024), blk, 0, stream>>>(Qp, Kp, Vp, ctx);

    // output projection -> fp32 result
    mha_proj_out<<<pgrid, blk, 0, stream>>>(ctx, wob, bo, out);
}
